// SelfAttentionHead_15333033247071
// MI455X (gfx1250) — matrix operand order repaired and checked by reading, not compiled, not run on hardware
//
#include <hip/hip_runtime.h>

// ---------------------------------------------------------------------------
// Fused causal self-attention head for MI455X (gfx1250, wave32, WMMA).
// One workgroup (256 threads = 8 waves) handles one batch element b:
//   x_b [128x384] f32 -> bf16 LDS (read from HBM exactly once)
//   K/Q/V = x_b @ W^T via v_wmma_f32_16x16x32_bf16, kept in LDS (bf16)
//   S = Q K^T, causal mask + softmax in WMMA C-layout registers
//   O = P V, normalized by per-row softmax sum at the end, stored f32
// LDS footprint ~240KB -> 1 WG per WGP (320KB LDS on CDNA5).
// All fragment loads are 16B-aligned ds_load_b128 (2 per fragment).
// ---------------------------------------------------------------------------

typedef __attribute__((ext_vector_type(16))) __bf16 v16bf;
typedef __attribute__((ext_vector_type(8)))  float  v8f;

#define TSEQ 128
#define CEMB 384
#define HS   64
#define XS_STRIDE 392   // 384 + 8 pad (ushorts); 784B = 49*16 -> rows 16B aligned
#define W_STRIDE  392
#define KQ_STRIDE 72    // 144B = 9*16
#define VT_STRIDE 136   // 272B = 17*16
#define P_STRIDE  136

union FragBF {
    v16bf v;
    uint4 q[2];
};

__device__ __forceinline__ unsigned short f2bf(float f) {
    unsigned int u = __float_as_uint(f);
    u += 0x7FFFu + ((u >> 16) & 1u);   // round-to-nearest-even bf16
    return (unsigned short)(u >> 16);
}

__device__ __forceinline__ unsigned int pk2bf(float lo, float hi) {
    return (unsigned int)f2bf(lo) | ((unsigned int)f2bf(hi) << 16);
}

// A-matrix 16x32 bf16 fragment (ISA 7.12.2 layout):
// lane half=0 (lanes 0-15): row = lane, K pairs {0,2,4,6,16,18,20,22}
// lane half=1 (lanes 16-31): row = lane-16, same pattern + 8
// -> two contiguous 16B runs: halves [0..7] and [16..23] after hf*8 shift.
__device__ __forceinline__ void load_frag_a(FragBF& f, const unsigned short* base,
                                            int stride, int r0, int k0, int lane) {
    const int hf = lane >> 4;
    const int r  = lane & 15;
    const unsigned short* p = base + (r0 + r) * stride + k0 + hf * 8;
    f.q[0] = *(const uint4*)(p);        // K halves 0..7
    f.q[1] = *(const uint4*)(p + 16);   // K halves 16..23
}

// B-matrix 32x16 bf16 fragment (ISA sparse-B 64x16 pattern scaled to K=32):
// lanes 0-15 hold K=0..7,16..23; lanes 16-31 hold K=8..15,24..31 (matches A).
// element (k, col) stored k-contiguously at base[(n0+col)*stride + k0 + k]
// -> two 16B runs per lane: two ds_load_b128.
__device__ __forceinline__ void load_frag_b(FragBF& f, const unsigned short* base,
                                            int stride, int n0, int k0, int lane) {
    const int hf = lane >> 4;
    const int c  = lane & 15;
    const unsigned short* p = base + (n0 + c) * stride + k0 + hf * 8;
    f.q[0] = *(const uint4*)(p);
    f.q[1] = *(const uint4*)(p + 16);
}

__global__ __launch_bounds__(256) void attn_head_kernel(
    const float* __restrict__ x,  const float* __restrict__ Wk,
    const float* __restrict__ Wq, const float* __restrict__ Wv,
    float* __restrict__ out)
{
    extern __shared__ unsigned short smem[];
    unsigned short* xs   = smem;                           // 128 x 392
    unsigned short* wbuf = xs   + TSEQ * XS_STRIDE;        //  64 x 392
    unsigned short* Ks   = wbuf + HS   * W_STRIDE;         // 128 x 72  (row s, col h)
    unsigned short* Qs   = Ks   + TSEQ * KQ_STRIDE;        // 128 x 72
    unsigned short* Vt   = Qs   + TSEQ * KQ_STRIDE;        //  64 x 136 (row h, col s)
    unsigned short* pbuf = Vt   + HS   * VT_STRIDE;        // 8 x 16 x 136

    const int b    = blockIdx.x;
    const int tid  = threadIdx.x;
    const int lane = tid & 31;
    const int wid  = tid >> 5;
    const int m0   = wid * 16;
    const int hf   = lane >> 4;     // which half-wave
    const int cb   = lane & 15;     // column-within-tile

    // ---- Phase 1: stream x_b into LDS as bf16 (single HBM read of x) ----
    const float* xb = x + (size_t)b * TSEQ * CEMB;
#pragma unroll 4
    for (int i = tid; i < TSEQ * CEMB / 4; i += 256) {
        float4 v = ((const float4*)xb)[i];
        int flat = i * 4;
        int t = flat / CEMB;
        int c = flat - t * CEMB;          // CEMB % 4 == 0 -> no row straddle
        uint2 pk;
        pk.x = pk2bf(v.x, v.y);
        pk.y = pk2bf(v.z, v.w);
        *(uint2*)(xs + t * XS_STRIDE + c) = pk;   // 8B-aligned ds_store_b64
    }

    const v8f vzero = {0.f,0.f,0.f,0.f,0.f,0.f,0.f,0.f};

    // ---- Phase 2: projections K, Q, V (reuse wbuf per weight matrix) ----
    const float* Ws[3] = {Wk, Wq, Wv};
    for (int p = 0; p < 3; ++p) {
        __syncthreads();                   // previous users of wbuf done
        const float* W = Ws[p];
        for (int i = tid; i < HS * CEMB / 4; i += 256) {
            float4 v = ((const float4*)W)[i];
            int flat = i * 4;
            int h = flat / CEMB;
            int c = flat - h * CEMB;
            uint2 pk;
            pk.x = pk2bf(v.x, v.y);
            pk.y = pk2bf(v.z, v.w);
            *(uint2*)(wbuf + h * W_STRIDE + c) = pk;
        }
        __syncthreads();

        v8f acc[4] = {vzero, vzero, vzero, vzero};
#pragma unroll
        for (int kt = 0; kt < CEMB / 32; ++kt) {
            FragBF a;
            load_frag_a(a, xs, XS_STRIDE, m0, kt * 32, lane);
#pragma unroll
            for (int nt = 0; nt < 4; ++nt) {
                FragBF bm;
                load_frag_b(bm, wbuf, W_STRIDE, nt * 16, kt * 32, lane);
                acc[nt] = __builtin_amdgcn_wmma_f32_16x16x32_bf16(
                    false, a.v, false, bm.v, (short)0, acc[nt], false, false);
            }
        }
        // scatter C-layout tile into LDS as bf16
#pragma unroll
        for (int nt = 0; nt < 4; ++nt) {
#pragma unroll
            for (int r = 0; r < 8; ++r) {
                int row = m0 + (hf ? 8 + r : r);
                int col = nt * 16 + cb;
                unsigned short bv = f2bf(acc[nt][r]);
                if      (p == 0) Ks[row * KQ_STRIDE + col] = bv;
                else if (p == 1) Qs[row * KQ_STRIDE + col] = bv;
                else             Vt[col * VT_STRIDE + row] = bv;  // V transposed
            }
        }
    }
    __syncthreads();

    // ---- Phase 3: S = Q K^T (each wave: its 16 rows x all 128 keys) ----
    FragBF qa0, qa1;
    load_frag_a(qa0, Qs, KQ_STRIDE, m0, 0,  lane);
    load_frag_a(qa1, Qs, KQ_STRIDE, m0, 32, lane);
    v8f sacc[8];
#pragma unroll
    for (int nt = 0; nt < 8; ++nt) {
        v8f c = vzero;
        FragBF bk;
        load_frag_b(bk, Ks, KQ_STRIDE, nt * 16, 0, lane);
        c = __builtin_amdgcn_wmma_f32_16x16x32_bf16(false, qa0.v, false, bk.v,
                                                    (short)0, c, false, false);
        load_frag_b(bk, Ks, KQ_STRIDE, nt * 16, 32, lane);
        c = __builtin_amdgcn_wmma_f32_16x16x32_bf16(false, qa1.v, false, bk.v,
                                                    (short)0, c, false, false);
        sacc[nt] = c;
    }

    // ---- Phase 4: causal mask + softmax in C-layout registers ----
    // Row m lives across 16 lanes of one half-wave (n = 16*t + cb), so
    // max/sum reduce with xor-shuffles 1,2,4,8 (stay within the half).
    unsigned short* pw = pbuf + wid * 16 * P_STRIDE;
    float rinv[8];
    const float kexp = 0.073621599f;   // (384^-0.5) * log2(e)
#pragma unroll
    for (int r = 0; r < 8; ++r) {
        const int rowl = hf ? 8 + r : r;
        const int row  = m0 + rowl;
        float sv[8];
        float mx = -3.0e38f;
#pragma unroll
        for (int t = 0; t < 8; ++t) {
            int col = t * 16 + cb;
            float s = sacc[t][r];
            s = (col <= row) ? s : -3.0e38f;   // causal mask
            sv[t] = s;
            mx = fmaxf(mx, s);
        }
        mx = fmaxf(mx, __shfl_xor(mx, 1, 32));
        mx = fmaxf(mx, __shfl_xor(mx, 2, 32));
        mx = fmaxf(mx, __shfl_xor(mx, 4, 32));
        mx = fmaxf(mx, __shfl_xor(mx, 8, 32));
        float sum = 0.f;
#pragma unroll
        for (int t = 0; t < 8; ++t) {
            float pe = exp2f((sv[t] - mx) * kexp);   // masked -> exp2(-huge) = 0
            sum += pe;
            pw[rowl * P_STRIDE + t * 16 + cb] = f2bf(pe);  // unnormalized P
        }
        sum += __shfl_xor(sum, 1, 32);
        sum += __shfl_xor(sum, 2, 32);
        sum += __shfl_xor(sum, 4, 32);
        sum += __shfl_xor(sum, 8, 32);
        rinv[r] = 1.0f / sum;          // valid for this half-wave's rows
    }
    __syncthreads();   // make all P stores visible before A-fragment loads

    // ---- Phase 5: O = P V, normalize, store f32 ----
    FragBF pa[4];
#pragma unroll
    for (int kt = 0; kt < 4; ++kt)
        load_frag_a(pa[kt], pw, P_STRIDE, 0, kt * 32, lane);

    float* ob = out + (size_t)b * TSEQ * HS;
#pragma unroll
    for (int nt = 0; nt < 4; ++nt) {
        v8f c = vzero;
#pragma unroll
        for (int kt = 0; kt < 4; ++kt) {
            FragBF vb;
            load_frag_b(vb, Vt, VT_STRIDE, nt * 16, kt * 32, lane);
            c = __builtin_amdgcn_wmma_f32_16x16x32_bf16(
                false, pa[kt].v, false, vb.v, (short)0, c, false, false);
        }
#pragma unroll
        for (int r = 0; r < 8; ++r) {
            int row = m0 + (hf ? 8 + r : r);
            int col = nt * 16 + cb;
            ob[row * HS + col] = c[r] * rinv[r];
        }
    }
}

extern "C" void kernel_launch(void* const* d_in, const int* in_sizes, int n_in,
                              void* d_out, int out_size, void* d_ws, size_t ws_size,
                              hipStream_t stream) {
    const float* x  = (const float*)d_in[0];
    const float* Wk = (const float*)d_in[1];
    const float* Wq = (const float*)d_in[2];
    const float* Wv = (const float*)d_in[3];
    float* out = (float*)d_out;

    const int B = in_sizes[0] / (TSEQ * CEMB);   // 1024

    // LDS: (128*392 + 64*392 + 2*128*72 + 64*136 + 8*16*136) ushorts = 239,616 B
    const size_t smem_bytes =
        (size_t)(TSEQ * XS_STRIDE + HS * W_STRIDE + 2 * TSEQ * KQ_STRIDE +
                 HS * VT_STRIDE + 8 * 16 * P_STRIDE) * sizeof(unsigned short);

    attn_head_kernel<<<dim3(B), dim3(256), smem_bytes, stream>>>(x, Wk, Wq, Wv, out);
}
